// Attention_5162550689965
// MI455X (gfx1250) — matrix operand order repaired and checked by reading, not compiled, not run on hardware
//
#include <hip/hip_runtime.h>
#include <hip/hip_bf16.h>
#include <math.h>

// ---------------------------------------------------------------------------
// Llama-style attention layer for MI455X (gfx1250), f16 WMMA everywhere,
// with TDM (tensor_load_to_lds) + async-to-LDS staging on the CDNA5 paths.
// B=1, S=2048, HIDDEN=4096, 32 heads, 8 KV heads, head_dim=128.
// ---------------------------------------------------------------------------

typedef __attribute__((ext_vector_type(8)))  _Float16 v8h;
typedef __attribute__((ext_vector_type(16))) _Float16 v16h;
typedef __attribute__((ext_vector_type(8)))  float    v8f;
typedef __attribute__((ext_vector_type(4)))  unsigned int u32x4;
typedef __attribute__((ext_vector_type(8)))  int      i32x8;
typedef __attribute__((ext_vector_type(4)))  int      i32x4;

#define S_LEN 2048
#define HID   4096
#define KVW   1024   // 8 kv heads * 128
#define NH    32
#define HD    128

// ---- CDNA5 feature detection ----------------------------------------------
#if defined(__has_builtin)
#  if __has_builtin(__builtin_amdgcn_tensor_load_to_lds) && \
      __has_builtin(__builtin_amdgcn_s_wait_tensorcnt)
#    define USE_TDM 1
#  endif
#endif
#define USE_ASYNC_LDS 1  // raw gfx1250 mnemonics via inline asm

// per-lane 16B global -> LDS copy, tracked with ASYNCcnt
static __device__ __forceinline__ void async_ld128(const _Float16* g,
                                                   const _Float16* l) {
  unsigned lds = (unsigned)(unsigned long long)l;       // LDS byte address
  unsigned long long ga = (unsigned long long)g;        // 64-bit global addr
  asm volatile("global_load_async_to_lds_b128 %0, %1, off"
               :: "v"(lds), "v"(ga) : "memory");
}
static __device__ __forceinline__ void wait_async0() {
  asm volatile("s_wait_asynccnt 0x0" ::: "memory");
}

#if defined(USE_TDM)
// Issue a 2D TDM tile load: tile_d x tile_r f16 elements, global row stride
// row_stride (elements), LDS rows padded 256B data + 16B pad (-> [r][136]).
static __device__ __forceinline__ void tdm_load_2d(const _Float16* g,
                                                   unsigned lds_off,
                                                   int tile_d, int tile_r,
                                                   int row_stride) {
  const unsigned long long ga = (unsigned long long)g;
  u32x4 g0;
  g0[0] = 1u;                                      // count=1, user mode
  g0[1] = lds_off;                                 // LDS byte address
  g0[2] = (unsigned)(ga & 0xffffffffu);            // global_addr[31:0]
  g0[3] = (unsigned)((ga >> 32) & 0x01ffffffu) | (2u << 30);  // [56:32]|type=2
  i32x8 g1;
  // data_size=2B, pad_enable, pad_interval=64 DWORDs (256B), pad_amount=4 DW
  g1[0] = (1 << 16) | (1 << 20) | (5 << 22) | (3 << 25);
  g1[1] = (tile_d & 0xffff) << 16;                 // tensor_dim0 lo16
  g1[2] = (4096 << 16);                            // dim0 hi=0 | tensor_dim1
  g1[3] = (tile_d & 0xffff) << 16;                 // dim1 hi=0 | tile_dim0
  g1[4] = tile_r & 0xffff;                         // tile_dim1 | tile_dim2=0
  g1[5] = row_stride;                              // tensor_dim0_stride lo32
  g1[6] = 0;                                       // stride0 hi | stride1 lo
  g1[7] = 0;                                       // stride1 hi
  i32x4 gz = {0, 0, 0, 0};
#if __clang_major__ >= 23
  i32x8 gz8 = {};
  __builtin_amdgcn_tensor_load_to_lds(g0, g1, gz, gz, gz8, 0);
#else
  __builtin_amdgcn_tensor_load_to_lds(g0, g1, gz, gz, 0);
#endif
}
#endif

static __device__ __forceinline__ v16h cat8(v8h a, v8h b) {
  return __builtin_shufflevector(a, b, 0, 1, 2, 3, 4, 5, 6, 7,
                                       8, 9, 10, 11, 12, 13, 14, 15);
}

static __device__ __forceinline__ v8f wmma16(v16h a, v16h b, v8f c) {
  // D = A(16x32 f16) * B(32x16 f16) + C(16x16 f32)
  return __builtin_amdgcn_wmma_f32_16x16x32_f16(false, a, false, b,
                                                (short)0, c, false, false);
}

// ------------------------------- convert -----------------------------------
__global__ void cvt_f32_f16(const float* __restrict__ in,
                            _Float16* __restrict__ out, int n) {
  int i = blockIdx.x * blockDim.x + threadIdx.x;
  if (i < n) out[i] = (_Float16)in[i];
}

// ------------------------------- RoPE --------------------------------------
__global__ void rope_kernel(_Float16* __restrict__ buf,
                            const int* __restrict__ pos, int W, int npairs) {
  int i = blockIdx.x * blockDim.x + threadIdx.x;
  if (i >= npairs) return;
  int pw   = W >> 1;
  int s    = i / pw;
  int rem  = i - s * pw;
  int head = rem >> 6;   // 64 pairs per head
  int pi   = rem & 63;
  float invf = __powf(10000.0f, -(float)(2 * pi) * (1.0f / 128.0f));
  float ang  = (float)pos[s] * invf;
  float sn, cs;
  __sincosf(ang, &sn, &cs);
  size_t off = (size_t)s * W + head * 128 + 2 * pi;
  float re = (float)buf[off], im = (float)buf[off + 1];
  buf[off]     = (_Float16)(re * cs - im * sn);
  buf[off + 1] = (_Float16)(re * sn + im * cs);
}

// ------------------------------- GEMM --------------------------------------
// C[M,N] = A[M,K] * B[K,N]; A,B f16 row-major; OutT = f16 or f32.
// Block tile 128x128x32, 8 waves; each wave computes 64x32 via 4x2 WMMA accs.
template <typename OutT>
__global__ void __launch_bounds__(256)
gemm_wmma(const _Float16* __restrict__ A, const _Float16* __restrict__ B,
          OutT* __restrict__ C, int M, int N, int K) {
  __shared__ __align__(16) _Float16 As[128][40];    // row-major A tile
  __shared__ __align__(16) _Float16 Bst[128][40];   // transposed B tile: [n][k]

  const int tid  = threadIdx.x;
  const int wave = tid >> 5;
  const int lane = tid & 31;
  const int ln   = lane & 15;
  const int hi   = lane >> 4;
  const int wm   = wave >> 2;  // 0..1
  const int wn   = wave & 3;   // 0..3
  const int bm   = blockIdx.y * 128;
  const int bn   = blockIdx.x * 128;

  v8f acc[4][2] = {};

  const int ar = tid >> 1, ac = (tid & 1) * 16;  // A-tile cooperative load
  const int bk = tid >> 3, bc = (tid & 7) * 16;  // B-tile cooperative load

  for (int k0 = 0; k0 < K; k0 += 32) {
    {
      const _Float16* pa = A + (size_t)(bm + ar) * K + k0 + ac;
#if defined(USE_ASYNC_LDS)
      async_ld128(pa, &As[ar][ac]);          // ASYNCcnt-tracked LDS fill
      async_ld128(pa + 8, &As[ar][ac + 8]);
#else
      *(v8h*)&As[ar][ac]     = *(const v8h*)pa;
      *(v8h*)&As[ar][ac + 8] = *(const v8h*)(pa + 8);
#endif
      const _Float16* pb = B + (size_t)(k0 + bk) * N + bn + bc;
      v8h b0 = *(const v8h*)pb;
      v8h b1 = *(const v8h*)(pb + 8);
#pragma unroll
      for (int e = 0; e < 8; ++e) {
        Bst[bc + e][bk]     = b0[e];
        Bst[bc + 8 + e][bk] = b1[e];
      }
    }
#if defined(USE_ASYNC_LDS)
    wait_async0();
#endif
    __syncthreads();

    if (k0 + 32 < K)  // warm caches for next A tile -> global_prefetch_b8
      __builtin_prefetch(A + (size_t)(bm + ar) * K + k0 + 32 + ac, 0, 0);

    // A fragments: lane row m = ln; K half base 0/8 (+16 upper chunk)
    v16h af[4];
#pragma unroll
    for (int i = 0; i < 4; ++i) {
      const int row = wm * 64 + i * 16 + ln;
      const int kk  = hi ? 8 : 0;
      af[i] = cat8(*(const v8h*)&As[row][kk], *(const v8h*)&As[row][kk + 16]);
    }
    // B fragments: lane col n = ln; K half base 0/8 (+16 upper chunk)
    v16h bf[2];
#pragma unroll
    for (int j = 0; j < 2; ++j) {
      const int col = wn * 32 + j * 16 + ln;
      const int kk  = hi ? 8 : 0;
      bf[j] = cat8(*(const v8h*)&Bst[col][kk], *(const v8h*)&Bst[col][kk + 16]);
    }
#pragma unroll
    for (int i = 0; i < 4; ++i)
#pragma unroll
      for (int j = 0; j < 2; ++j)
        acc[i][j] = wmma16(af[i], bf[j], acc[i][j]);
    __syncthreads();
  }

#pragma unroll
  for (int i = 0; i < 4; ++i)
#pragma unroll
    for (int j = 0; j < 2; ++j)
#pragma unroll
      for (int r = 0; r < 8; ++r) {
        const int row = bm + wm * 64 + i * 16 + r + 8 * hi;
        const int col = bn + wn * 32 + j * 16 + ln;
        C[(size_t)row * N + col] = (OutT)acc[i][j][r];
      }
}

// --------------------------- flash attention -------------------------------
// Grid: (S/128 q-tiles, 32 heads). 8 waves; wave w owns q rows [w*16, w*16+16).
// Streams 32-key blocks with online softmax; GQA maps head h -> kv head h>>2.
__global__ void __launch_bounds__(256)
flash_attn(const _Float16* __restrict__ Q, const _Float16* __restrict__ Kg,
           const _Float16* __restrict__ Vg, _Float16* __restrict__ O) {
  __shared__ __align__(16) _Float16 Ks[32][136];     // [key][d] row-major
  __shared__ __align__(16) _Float16 Vt[128][40];     // [d][key] transposed
  __shared__ __align__(16) _Float16 Pst[8][16][40];  // per-wave P relayout

  const int h    = blockIdx.y;
  const int kvh  = h >> 2;
  const int qt   = blockIdx.x;
  const int tid  = threadIdx.x;
  const int wave = tid >> 5;
  const int lane = tid & 31;
  const int ln   = lane & 15;
  const int hi   = lane >> 4;
  const int qrow = qt * 128 + wave * 16;
  const float scale = 0.08838834764831845f;  // 1/sqrt(128)

  // Q fragments held in registers for the whole key sweep (16 rows x 128 d)
  v16h qf[4];
#pragma unroll
  for (int dc = 0; dc < 4; ++dc) {
    const int kk = hi ? 8 : 0;
    const _Float16* qp = Q + (size_t)(qrow + ln) * HID + h * HD + dc * 32 + kk;
    qf[dc] = cat8(*(const v8h*)qp, *(const v8h*)(qp + 16));
  }

  float mrow[8], lrow[8];
  v8f o[8] = {};
#pragma unroll
  for (int r = 0; r < 8; ++r) { mrow[r] = -__builtin_inff(); lrow[r] = 0.0f; }

  const int ldk = tid >> 3, ldc = (tid & 7) * 16;
  const int nkb = 4 * (qt + 1);  // causal: only key blocks up to q-tile end

  for (int kb = 0; kb < nkb; ++kb) {
    {  // stage K (TDM / async) and V (transposed, sync)
      const size_t base = (size_t)(kb * 32 + ldk) * KVW + kvh * HD + ldc;
#if defined(USE_TDM)
      if (wave == 0) {
        // 32 rows x 128 f16, global row stride 1024 elems, LDS rows padded
        // to 272B -> lands exactly in Ks[32][136]. TENSORcnt-tracked DMA.
        tdm_load_2d(Kg + (size_t)(kb * 32) * KVW + kvh * HD,
                    (unsigned)(unsigned long long)(&Ks[0][0]), HD, 32, KVW);
      }
#else
      async_ld128(Kg + base, &Ks[ldk][ldc]);
      async_ld128(Kg + base + 8, &Ks[ldk][ldc + 8]);
#endif
      v8h v0 = *(const v8h*)(Vg + base);
      v8h v1 = *(const v8h*)(Vg + base + 8);
#pragma unroll
      for (int e = 0; e < 8; ++e) {
        Vt[ldc + e][ldk]     = v0[e];
        Vt[ldc + 8 + e][ldk] = v1[e];
      }
#if defined(USE_TDM)
      if (wave == 0) __builtin_amdgcn_s_wait_tensorcnt(0);
#else
      wait_async0();
#endif
    }
    __syncthreads();

    // scores: two 16-key sub-tiles, accumulate over head_dim (4 x K=32)
    v8f s[2] = {};
#pragma unroll
    for (int t = 0; t < 2; ++t) {
      const int krow = t * 16 + ln;
      const int kk   = hi ? 8 : 0;
#pragma unroll
      for (int dc = 0; dc < 4; ++dc) {
        v16h bf = cat8(*(const v8h*)&Ks[krow][dc * 32 + kk],
                       *(const v8h*)&Ks[krow][dc * 32 + kk + 16]);
        s[t] = wmma16(qf[dc], bf, s[t]);
      }
    }

    // online softmax (rows split across the two 16-lane halves)
    float sv[2][8], rmax[8], rsum[8], corr[8], p[2][8];
#pragma unroll
    for (int r = 0; r < 8; ++r) rmax[r] = -__builtin_inff();
#pragma unroll
    for (int t = 0; t < 2; ++t)
#pragma unroll
      for (int r = 0; r < 8; ++r) {
        float v = s[t][r] * scale;
        const int keyg = kb * 32 + t * 16 + ln;
        const int qg   = qrow + r + 8 * hi;
        if (keyg > qg) v = -__builtin_inff();  // causal mask
        sv[t][r] = v;
        rmax[r]  = fmaxf(rmax[r], v);
      }
#pragma unroll
    for (int r = 0; r < 8; ++r) {
#pragma unroll
      for (int m = 1; m < 16; m <<= 1)
        rmax[r] = fmaxf(rmax[r], __shfl_xor(rmax[r], m, 32));
      const float mn = fmaxf(mrow[r], rmax[r]);
      corr[r] = __expf(mrow[r] - mn);
      mrow[r] = mn;
      rsum[r] = 0.0f;
    }
#pragma unroll
    for (int t = 0; t < 2; ++t)
#pragma unroll
      for (int r = 0; r < 8; ++r) {
        const float pv = __expf(sv[t][r] - mrow[r]);
        p[t][r] = pv;
        rsum[r] += pv;
      }
#pragma unroll
    for (int r = 0; r < 8; ++r) {
#pragma unroll
      for (int m = 1; m < 16; m <<= 1)
        rsum[r] += __shfl_xor(rsum[r], m, 32);
      lrow[r] = lrow[r] * corr[r] + rsum[r];
    }
#pragma unroll
    for (int d = 0; d < 8; ++d)
#pragma unroll
      for (int r = 0; r < 8; ++r)
        o[d][r] *= corr[r];

    // relayout P from C-layout to A-fragment layout via per-wave LDS patch
#pragma unroll
    for (int t = 0; t < 2; ++t)
#pragma unroll
      for (int r = 0; r < 8; ++r)
        Pst[wave][r + 8 * hi][t * 16 + ln] = (_Float16)p[t][r];
    __syncthreads();

    {  // O += P(16x32) * V(32x16 per d-tile)
      const int kkA = hi ? 8 : 0;
      v16h af = cat8(*(const v8h*)&Pst[wave][ln][kkA],
                     *(const v8h*)&Pst[wave][ln][kkA + 16]);
#pragma unroll
      for (int dt = 0; dt < 8; ++dt) {
        const int kkB = hi ? 8 : 0;
        v16h bf = cat8(*(const v8h*)&Vt[dt * 16 + ln][kkB],
                       *(const v8h*)&Vt[dt * 16 + ln][kkB + 16]);
        o[dt] = wmma16(af, bf, o[dt]);
      }
    }
    __syncthreads();
  }

#pragma unroll
  for (int dt = 0; dt < 8; ++dt)
#pragma unroll
    for (int r = 0; r < 8; ++r) {
      const int qg = qrow + r + 8 * hi;
      O[(size_t)qg * HID + h * HD + dt * 16 + ln] =
          (_Float16)(o[dt][r] / lrow[r]);
    }
}

// ------------------------------- launch ------------------------------------
extern "C" void kernel_launch(void* const* d_in, const int* in_sizes, int n_in,
                              void* d_out, int out_size, void* d_ws,
                              size_t ws_size, hipStream_t stream) {
  (void)in_sizes; (void)n_in; (void)out_size; (void)ws_size;

  const float* x   = (const float*)d_in[0];
  // d_in[1] = attention_mask (all-true for this problem; causal mask applied)
  const int*   pos = (const int*)d_in[2];
  const float* wq  = (const float*)d_in[3];
  const float* wk  = (const float*)d_in[4];
  const float* wv  = (const float*)d_in[5];
  const float* wo  = (const float*)d_in[6];
  float* out = (float*)d_out;

  _Float16* p   = (_Float16*)d_ws;
  _Float16* Xh  = p; p += (size_t)S_LEN * HID;
  _Float16* Wqh = p; p += (size_t)HID * HID;
  _Float16* Wkh = p; p += (size_t)HID * KVW;
  _Float16* Wvh = p; p += (size_t)HID * KVW;
  _Float16* Woh = p; p += (size_t)HID * HID;
  _Float16* Qh  = p; p += (size_t)S_LEN * HID;
  _Float16* Kh  = p; p += (size_t)S_LEN * KVW;
  _Float16* Vh  = p; p += (size_t)S_LEN * KVW;
  _Float16* AOh = p; p += (size_t)S_LEN * HID;

  cvt_f32_f16<<<(S_LEN * HID) / 256, 256, 0, stream>>>(x, Xh, S_LEN * HID);
  cvt_f32_f16<<<(HID * HID) / 256, 256, 0, stream>>>(wq, Wqh, HID * HID);
  cvt_f32_f16<<<(HID * KVW) / 256, 256, 0, stream>>>(wk, Wkh, HID * KVW);
  cvt_f32_f16<<<(HID * KVW) / 256, 256, 0, stream>>>(wv, Wvh, HID * KVW);
  cvt_f32_f16<<<(HID * HID) / 256, 256, 0, stream>>>(wo, Woh, HID * HID);

  gemm_wmma<_Float16><<<dim3(HID / 128, S_LEN / 128), 256, 0, stream>>>(
      Xh, Wqh, Qh, S_LEN, HID, HID);
  gemm_wmma<_Float16><<<dim3(KVW / 128, S_LEN / 128), 256, 0, stream>>>(
      Xh, Wkh, Kh, S_LEN, KVW, HID);
  gemm_wmma<_Float16><<<dim3(KVW / 128, S_LEN / 128), 256, 0, stream>>>(
      Xh, Wvh, Vh, S_LEN, KVW, HID);

  rope_kernel<<<(S_LEN * HID / 2) / 256, 256, 0, stream>>>(
      Qh, pos, HID, S_LEN * HID / 2);
  rope_kernel<<<(S_LEN * KVW / 2) / 256, 256, 0, stream>>>(
      Kh, pos, KVW, S_LEN * KVW / 2);

  flash_attn<<<dim3(S_LEN / 128, NH), 256, 0, stream>>>(Qh, Kh, Vh, AOh);

  gemm_wmma<float><<<dim3(HID / 128, S_LEN / 128), 256, 0, stream>>>(
      AOh, Woh, out, S_LEN, HID, HID);
}
